// NoisyTopkRouter_8504035246114
// MI455X (gfx1250) — matrix operand order repaired and checked by reading, not compiled, not run on hardware
//
#include <hip/hip_runtime.h>
#include <cstdint>

// NoisyTopkRouter (eval): h = relu(x@W1+b1); logits = h@W2+b2; top-2 + sparse softmax.
// GEMM1 (137 GFLOP) is matrix-compute bound on MI455X -> bf16 WMMA with
// LDS double-buffered, register-prefetched pipeline (1 barrier per K-step),
// sched_barrier-pinned prefetch issue so WMMAs cover the load latency.
// GEMM2 stages the bf16 h tiles via GLOBAL_LOAD_ASYNC_TO_LDS (ASYNCcnt path).

static constexpr int IN_F  = 4096;
static constexpr int HID   = 1024;
static constexpr int NE    = 64;
static constexpr int NROWS = 16384;

typedef __attribute__((ext_vector_type(16))) __bf16 v16bf;
typedef __attribute__((ext_vector_type(8)))  __bf16 v8bf;
typedef __attribute__((ext_vector_type(2)))  __bf16 v2bf;
typedef __attribute__((ext_vector_type(8)))  float  v8f;
typedef __attribute__((ext_vector_type(4)))  float  v4f;

#if defined(__AMDGCN__) && __has_builtin(__builtin_amdgcn_global_load_async_to_lds_b128)
#define HAVE_ASYNC_COPY 1
// builtin signature (from clang diagnostic): (global int4*, shared int4*, imm, imm)
typedef int v4i_ __attribute__((vector_size(16)));
typedef __attribute__((address_space(1))) v4i_ as1_v4i;
typedef __attribute__((address_space(3))) v4i_ as3_v4i;
__device__ __forceinline__ void wait_async0() {
#if __has_builtin(__builtin_amdgcn_s_wait_asynccnt)
  __builtin_amdgcn_s_wait_asynccnt(0);
#else
  asm volatile("s_wait_asynccnt 0" ::: "memory");
#endif
}
#else
#define HAVE_ASYNC_COPY 0
#endif

__device__ __forceinline__ void sched_fence() {
#if __has_builtin(__builtin_amdgcn_sched_barrier)
  __builtin_amdgcn_sched_barrier(0);   // pin prefetch issue point
#endif
}

__device__ __forceinline__ v16bf cat16(v8bf lo, v8bf hi) {
  v16bf r;
#pragma unroll
  for (int i = 0; i < 8; ++i) { r[i] = lo[i]; r[i + 8] = hi[i]; }
  return r;
}

// ---------------------------------------------------------------------------
// Kernel 1: h = relu(x @ W1 + b1), stored bf16 in workspace.
// Block tile 128x128, K-step 32 (= one WMMA K). 8 waves: 4 row-groups x
// 2 col-groups, each wave owns 2x4 WMMA 16x16 tiles. LDS double-buffered;
// next tile prefetched into registers while current tile feeds the WMMAs.
// ---------------------------------------------------------------------------
__global__ __launch_bounds__(256)
void gemm1_relu_bf16(const float* __restrict__ x, const float* __restrict__ W1,
                     const float* __restrict__ b1, __bf16* __restrict__ h)
{
  __shared__ __bf16 As[2][128][40];   // x tile, row-major  [m][k], +8 pad
  __shared__ __bf16 Bs[2][128][40];   // W1 tile, TRANSPOSED [n][k], +8 pad

  const int tid  = threadIdx.x;
  const int lane = tid & 31;
  const int w    = tid >> 5;
  const int wm   = w & 3;            // 32-row group
  const int wn   = w >> 2;           // 64-col group
  const int r0   = blockIdx.y * 128;
  const int c0   = blockIdx.x * 128;

  // wave32 WMMA fragment indexing (ISA 7.12.2)
  const int mrow = lane & 15;        // A: row within 16x32 tile
  const int half = lane >> 4;
  const int kA0  = half ? 8 : 0;     // A: lanes 0-15 hold K 0-7/16-23; 16-31 hold 8-15/24-31
  const int ncol = lane & 15;        // B: column within 32x16 tile
  const int kB   = half << 4;        // B: same K order as A (lanes 0-15 K 0-7/16-23; 16-31 K 8-15/24-31)

  v8f acc[2][4] = {};

  const int arow = tid >> 1, acol = (tid & 1) << 4;   // x tile: 128 rows, 16 f32/thr
  const int bk0  = (tid & 15) << 1;                   // W1 tile: 2 k-rows x 8 cols/thr
  const int bn0  = (tid >> 4) << 3;

  v4f a0, a1, a2, a3;       // prefetch registers: x tile slice
  v4f q0, q1, q2, q3;       // prefetch registers: W1 slice (q0,q1: row k0; q2,q3: row k0+1)

  auto loadA = [&](int kt) {
    const float* src = x + (size_t)(r0 + arow) * IN_F + kt + acol;
    a0 = *(const v4f*)(src + 0);  a1 = *(const v4f*)(src + 4);
    a2 = *(const v4f*)(src + 8);  a3 = *(const v4f*)(src + 12);
  };
  auto loadB = [&](int kt) {
    const float* src = W1 + (size_t)(kt + bk0) * HID + c0 + bn0;
    q0 = *(const v4f*)(src);        q1 = *(const v4f*)(src + 4);
    q2 = *(const v4f*)(src + HID);  q3 = *(const v4f*)(src + HID + 4);
  };
  auto stage = [&](int buf) {           // convert prefetch regs -> LDS[buf]
    v8bf lo, hi8;
#pragma unroll
    for (int i = 0; i < 4; ++i) {
      lo[i]  = (__bf16)a0[i];  lo[i + 4]  = (__bf16)a1[i];
      hi8[i] = (__bf16)a2[i];  hi8[i + 4] = (__bf16)a3[i];
    }
    *(v8bf*)&As[buf][arow][acol]     = lo;
    *(v8bf*)&As[buf][arow][acol + 8] = hi8;
    // transposed B: (k0,k0+1) of one column are adjacent -> packed b32 stores
#pragma unroll
    for (int i = 0; i < 4; ++i) {
      v2bf p0; p0[0] = (__bf16)q0[i]; p0[1] = (__bf16)q2[i];
      v2bf p1; p1[0] = (__bf16)q1[i]; p1[1] = (__bf16)q3[i];
      *(v2bf*)&Bs[buf][bn0 + i    ][bk0] = p0;
      *(v2bf*)&Bs[buf][bn0 + i + 4][bk0] = p1;
    }
  };
  auto mma = [&](int buf) {
    v16bf afrag[2];
#pragma unroll
    for (int mt = 0; mt < 2; ++mt) {
      const __bf16* ap = &As[buf][wm * 32 + mt * 16 + mrow][0];
      afrag[mt] = cat16(*(const v8bf*)(ap + kA0), *(const v8bf*)(ap + kA0 + 16));
    }
#pragma unroll
    for (int nt = 0; nt < 4; ++nt) {
      const __bf16* bp = &Bs[buf][wn * 64 + nt * 16 + ncol][kA0];
      v16bf bfrag = cat16(*(const v8bf*)bp, *(const v8bf*)(bp + 16));
#pragma unroll
      for (int mt = 0; mt < 2; ++mt) {
        acc[mt][nt] = __builtin_amdgcn_wmma_f32_16x16x32_bf16(
            false, afrag[mt], false, bfrag, (short)0, acc[mt][nt], false, false);
      }
    }
  };

  // pipeline prologue: tile 0 staged, tile 1 in registers
  loadA(0);  loadB(0);
  stage(0);
  loadA(32); loadB(32);
  __syncthreads();

  int buf = 0;
  for (int kt = 0; kt < IN_F - 32; kt += 32) {
    stage(buf ^ 1);                         // tile kt+32: regs -> LDS[buf^1]
    int ktn = kt + 64;                      // issue tile kt+64 global loads now
    if (ktn >= IN_F) ktn = 0;               // (harmless dummy reload at the tail)
    loadA(ktn); loadB(ktn);
    sched_fence();                          // keep loads issued BEFORE the WMMA block
    mma(buf);                               // consume tile kt (covers load latency)
    __syncthreads();
    buf ^= 1;
  }
  mma(buf);                                 // last tile

  // epilogue: bias + ReLU -> bf16 h  (C/D layout: lanes 0-15 rows 0-7, 16-31 rows 8-15)
#pragma unroll
  for (int nt = 0; nt < 4; ++nt) {
    const int c = c0 + wn * 64 + nt * 16 + ncol;
    const float bias = b1[c];
#pragma unroll
    for (int mt = 0; mt < 2; ++mt) {
      const int rbase = r0 + wm * 32 + mt * 16 + half * 8;
#pragma unroll
      for (int r = 0; r < 8; ++r) {
        float v = acc[mt][nt][r] + bias;
        v = v > 0.0f ? v : 0.0f;
        h[(size_t)(rbase + r) * HID + c] = (__bf16)v;
      }
    }
  }
}

// ---------------------------------------------------------------------------
// Kernel 2: logits = h @ W2 + b2 (bf16 WMMA), then per-row top-2 + softmax.
// Block: 128 rows x 64 experts; 8 waves, each owns 16 rows x 4 N-tiles.
// h tiles are raw bf16 -> staged with GLOBAL_LOAD_ASYNC_TO_LDS (ASYNCcnt).
// ---------------------------------------------------------------------------
__global__ __launch_bounds__(256)
void gemm2_topk_softmax(const __bf16* __restrict__ h,
                        const float* __restrict__ W2, const float* __restrict__ b2,
                        float* __restrict__ out_router, int* __restrict__ out_idx)
{
  __shared__ __bf16 Hs[128][40];   // h tile [m][k]
  __shared__ __bf16 Ws[64][40];    // W2 tile transposed [n][k]
  __shared__ float  Ls[128][66];   // logits staging

  const int tid  = threadIdx.x;
  const int lane = tid & 31;
  const int w    = tid >> 5;       // 16-row group
  const int r0   = blockIdx.x * 128;

  const int mrow = lane & 15;
  const int half = lane >> 4;
  const int kA0  = half ? 8 : 0;
  const int ncol = lane & 15;
  const int kB   = half << 4;

  v8f acc[4] = {};

  const int arow = tid >> 1, acol = (tid & 1) << 4;   // h tile: 16 bf16/thr
  const int bk0  = (tid & 15) << 1;                   // W2 tile: 2 k-rows x 4 cols/thr
  const int bn0  = (tid >> 4) << 2;

  for (int kt = 0; kt < HID; kt += 32) {
    { // h tile is already bf16 -> async DMA to LDS (ASYNCcnt)
      const __bf16* src = h + (size_t)(r0 + arow) * HID + kt + acol;
#if HAVE_ASYNC_COPY
      __builtin_amdgcn_global_load_async_to_lds_b128(
          (as1_v4i*)src,       (as3_v4i*)&Hs[arow][acol],     0, 0);
      __builtin_amdgcn_global_load_async_to_lds_b128(
          (as1_v4i*)(src + 8), (as3_v4i*)&Hs[arow][acol + 8], 0, 0);
#else
      *(v8bf*)&Hs[arow][acol]     = *(const v8bf*)(src);
      *(v8bf*)&Hs[arow][acol + 8] = *(const v8bf*)(src + 8);
#endif
    }
    { // W2 tile f32 -> bf16, transposed with packed b32 stores
      const float* src = W2 + (size_t)(kt + bk0) * NE + bn0;
      v4f f0 = *(const v4f*)(src);
      v4f f1 = *(const v4f*)(src + NE);
#pragma unroll
      for (int i = 0; i < 4; ++i) {
        v2bf p; p[0] = (__bf16)f0[i]; p[1] = (__bf16)f1[i];
        *(v2bf*)&Ws[bn0 + i][bk0] = p;
      }
    }
#if HAVE_ASYNC_COPY
    wait_async0();
#endif
    __syncthreads();

    const __bf16* ap = &Hs[w * 16 + mrow][0];
    v16bf afrag = cat16(*(const v8bf*)(ap + kA0), *(const v8bf*)(ap + kA0 + 16));
#pragma unroll
    for (int nt = 0; nt < 4; ++nt) {
      const __bf16* bp = &Ws[nt * 16 + ncol][kA0];
      v16bf bfrag = cat16(*(const v8bf*)bp, *(const v8bf*)(bp + 16));
      acc[nt] = __builtin_amdgcn_wmma_f32_16x16x32_bf16(
          false, afrag, false, bfrag, (short)0, acc[nt], false, false);
    }
    __syncthreads();
  }

  // stage logits (+b2) to LDS
#pragma unroll
  for (int nt = 0; nt < 4; ++nt) {
    const int c = nt * 16 + ncol;
    const float bias = b2[c];
#pragma unroll
    for (int r = 0; r < 8; ++r)
      Ls[w * 16 + half * 8 + r][c] = acc[nt][r] + bias;
  }
  __syncthreads();

  // one thread per row: top-2, sparse softmax, emit probs + indices
  if (tid < 128) {
    const int row = r0 + tid;
    float m1 = -INFINITY, m2 = -INFINITY;
    int   i1 = 0, i2 = 0;
#pragma unroll
    for (int e = 0; e < NE; ++e) {
      float v = Ls[tid][e];
      if (v > m1)      { m2 = m1; i2 = i1; m1 = v; i1 = e; }
      else if (v > m2) { m2 = v;  i2 = e; }
    }
    const float e2  = __expf(m2 - m1);
    const float inv = 1.0f / (1.0f + e2);
    const float p1  = inv, p2 = e2 * inv;
    float* orow = out_router + (size_t)row * NE;
#pragma unroll
    for (int e = 0; e < NE; ++e) {
      float v = 0.0f;
      if (e == i1) v = p1;
      if (e == i2) v = p2;
      orow[e] = v;
    }
    out_idx[row * 2 + 0] = i1;
    out_idx[row * 2 + 1] = i2;
  }
}

extern "C" void kernel_launch(void* const* d_in, const int* in_sizes, int n_in,
                              void* d_out, int out_size, void* d_ws, size_t ws_size,
                              hipStream_t stream) {
  const float* x  = (const float*)d_in[0];
  const float* W1 = (const float*)d_in[1];
  const float* b1 = (const float*)d_in[2];
  const float* W2 = (const float*)d_in[3];
  const float* b2 = (const float*)d_in[4];

  __bf16* hbuf = (__bf16*)d_ws;                               // 16384x1024 bf16 = 32 MB
  float* out_router = (float*)d_out;                          // [N,64] f32
  int*   out_idx    = (int*)((float*)d_out + (size_t)NROWS * NE);  // [N,2] i32

  dim3 g1(HID / 128, NROWS / 128);
  gemm1_relu_bf16<<<g1, 256, 0, stream>>>(x, W1, b1, hbuf);
  gemm2_topk_softmax<<<NROWS / 128, 256, 0, stream>>>(hbuf, W2, b2, out_router, out_idx);
}
